// WTA_9526237462870
// MI455X (gfx1250) — matrix operand order repaired and checked by reading, not compiled, not run on hardware
//
#include <hip/hip_runtime.h>

// ---------------------------------------------------------------------------
// MI455X (gfx1250): all GEMMs via v_wmma_f32_16x16x32_f16, 32x32 output
// macro-tile per wave (2x2 fragment blocking).  Shared A macro-tiles are
// staged into LDS with global_load_async_to_lds_b128 (ASYNCcnt) and read
// back with ds_load_b128; B streams from global as b128 clauses.
// TN layout everywhere: A is [M][K] K-contiguous, B stored transposed [N][K].
// ---------------------------------------------------------------------------

typedef __attribute__((ext_vector_type(16))) _Float16 v16h;
typedef __attribute__((ext_vector_type(8)))  _Float16 v8h;
typedef __attribute__((ext_vector_type(8)))  float    v8f;
typedef __attribute__((ext_vector_type(4)))  unsigned int v4u;
typedef __attribute__((ext_vector_type(8)))  unsigned int v8u;

#define DIM   192
#define HEADS 4
#define CH    48            // channels per head
#define H     128
#define W     128
#define P     (H*W)         // 16384 pixels per channel
#define F     (CH*W)        // 6144 features per head-row
#define B     8
#define O3    (3*DIM)       // 576

// ---- workspace layout (bytes, all 256-aligned) ----------------------------
#define OFF_XT    ((size_t)0)                         // 50,331,648  f16 [B][P][DIM]   (reused as QH after K2)
#define OFF_WQ    ((size_t)50331648)                  // 221,184     f16 [O3][DIM]
#define OFF_WP    ((size_t)50552832)                  // 73,728      f16 [DIM][DIM]
#define OFF_NQ    ((size_t)50626560)                  // 16,384      f32 [B*HEADS*H]
#define OFF_NK    ((size_t)50642944)                  // 16,384      f32
#define OFF_QKV0  ((size_t)50659328)                  // 150,994,944 f16 [B][O3][P]    (dead after K3)
#define OFF_PO    OFF_QKV0                            // 50,331,648  f16 [B][P][DIM]   (alias)
#define OFF_SRAW  ((size_t)100990976)                 // 2,097,152   f32 [B*HEADS][H][H] (alias)
#define OFF_PA    ((size_t)103088128)                 // 1,048,576   f16 [B*HEADS][H][H] (alias)
#define OFF_KH    ((size_t)201654272)                 // 50,331,648  f16 [B*HEADS][H][F]
#define OFF_VT    ((size_t)251985920)                 // 50,331,648  f16 [B*HEADS][F][H]
#define OFF_QH    OFF_XT

// ---------------------------------------------------------------------------
// Fragment loaders per CDNA5 16-bit WMMA lane layouts (wave32).
//   A (16x32): lanes<16 hold K {0..7}+{16..23}, lanes>=16 K {8..15}+{24..31}
//   B (32x16): lanes<16 hold K {0..7}+{16..23}, lanes>=16 K {8..15}+{24..31}
// ---------------------------------------------------------------------------
__device__ __forceinline__ v16h ldfragA(const _Float16* __restrict__ p) {
  v8h lo = *(const v8h*)p, hi = *(const v8h*)(p + 16);
  v16h r;
#pragma unroll
  for (int i = 0; i < 8; ++i) { r[i] = lo[i]; r[i + 8] = hi[i]; }
  return r;
}
__device__ __forceinline__ v16h ldfragB(const _Float16* __restrict__ p) {
  v8h lo = *(const v8h*)p, hi = *(const v8h*)(p + 16);
  v16h r;
#pragma unroll
  for (int i = 0; i < 8; ++i) { r[i] = lo[i]; r[i + 8] = hi[i]; }
  return r;
}

// ---------------------------------------------------------------------------
// Async stage of a contiguous A macro-tile into LDS (16B chunks, 256 thr).
// Each wave waits its own ASYNCcnt, then the workgroup barrier guarantees
// the whole tile is visible.
// ---------------------------------------------------------------------------
__device__ __forceinline__ void stage_A_async(const _Float16* __restrict__ gsrc,
                                              _Float16* lds, int bytes, int tid) {
  unsigned ldsBase = (unsigned)(unsigned long long)lds;
  int chunks = bytes >> 4;
  for (int c = tid; c < chunks; c += 256) {
    unsigned la = ldsBase + ((unsigned)c << 4);
    unsigned long long ga = (unsigned long long)gsrc + ((unsigned long long)c << 4);
    asm volatile("global_load_async_to_lds_b128 %0, %1, off" :: "v"(la), "v"(ga) : "memory");
  }
  asm volatile("s_wait_asynccnt 0x0" ::: "memory");
  __syncthreads();
}

// ---------------------------------------------------------------------------
// 32x32 output macro-tile per wave (global A / global B)
// ---------------------------------------------------------------------------
__device__ __forceinline__ void gemm_block_2x2(const _Float16* __restrict__ A, int lda,
                                               const _Float16* __restrict__ BT, int ldb,
                                               int m0, int n0, int K, unsigned lane,
                                               v8f acc[2][2]) {
  const int half = (lane >> 4) & 1;
  const int l15  = lane & 15;
  const _Float16* a0 = A  + (size_t)(m0 + l15) * lda + half * 8;
  const _Float16* a1 = a0 + (size_t)16 * lda;
  const _Float16* b0 = BT + (size_t)(n0 + l15) * ldb + half * 8;
  const _Float16* b1 = b0 + (size_t)16 * ldb;
  acc[0][0] = {}; acc[0][1] = {}; acc[1][0] = {}; acc[1][1] = {};
  for (int k = 0; k < K; k += 32) {
    if (K > 1024 && k + 64 <= K) {          // long-K streams: global_prefetch_b8
      __builtin_prefetch(a0 + k + 64, 0, 3);
      __builtin_prefetch(a1 + k + 64, 0, 3);
      __builtin_prefetch(b0 + k + 64, 0, 3);
      __builtin_prefetch(b1 + k + 64, 0, 3);
    }
    v16h fa0 = ldfragA(a0 + k);
    v16h fa1 = ldfragA(a1 + k);
    v16h fb0 = ldfragB(b0 + k);
    v16h fb1 = ldfragB(b1 + k);
    acc[0][0] = __builtin_amdgcn_wmma_f32_16x16x32_f16(false, fa0, false, fb0, (short)0, acc[0][0], false, false);
    acc[0][1] = __builtin_amdgcn_wmma_f32_16x16x32_f16(false, fa0, false, fb1, (short)0, acc[0][1], false, false);
    acc[1][0] = __builtin_amdgcn_wmma_f32_16x16x32_f16(false, fa1, false, fb0, (short)0, acc[1][0], false, false);
    acc[1][1] = __builtin_amdgcn_wmma_f32_16x16x32_f16(false, fa1, false, fb1, (short)0, acc[1][1], false, false);
  }
}

// ---------------------------------------------------------------------------
// 32x32 output macro-tile per wave (LDS-staged A / global B).  As holds the
// 32-row A macro-tile (rows dense, lda == K), so m0 is implicitly 0.
// ---------------------------------------------------------------------------
__device__ __forceinline__ void gemm_block_2x2_sA(const _Float16* As, int lda,
                                                  const _Float16* __restrict__ BT, int ldb,
                                                  int n0, int K, unsigned lane,
                                                  v8f acc[2][2]) {
  const int half = (lane >> 4) & 1;
  const int l15  = lane & 15;
  const _Float16* a0 = As + (size_t)l15 * lda + half * 8;        // ds_load_b128 path
  const _Float16* a1 = a0 + (size_t)16 * lda;
  const _Float16* b0 = BT + (size_t)(n0 + l15) * ldb + half * 8;
  const _Float16* b1 = b0 + (size_t)16 * ldb;
  acc[0][0] = {}; acc[0][1] = {}; acc[1][0] = {}; acc[1][1] = {};
  for (int k = 0; k < K; k += 32) {
    v16h fa0 = ldfragA(a0 + k);
    v16h fa1 = ldfragA(a1 + k);
    v16h fb0 = ldfragB(b0 + k);
    v16h fb1 = ldfragB(b1 + k);
    acc[0][0] = __builtin_amdgcn_wmma_f32_16x16x32_f16(false, fa0, false, fb0, (short)0, acc[0][0], false, false);
    acc[0][1] = __builtin_amdgcn_wmma_f32_16x16x32_f16(false, fa0, false, fb1, (short)0, acc[0][1], false, false);
    acc[1][0] = __builtin_amdgcn_wmma_f32_16x16x32_f16(false, fa1, false, fb0, (short)0, acc[1][0], false, false);
    acc[1][1] = __builtin_amdgcn_wmma_f32_16x16x32_f16(false, fa1, false, fb1, (short)0, acc[1][1], false, false);
  }
}

// ---------------------------------------------------------------------------
// K0: weights f32 -> f16, zero norm accumulators
// ---------------------------------------------------------------------------
__global__ void k_prep(const float* __restrict__ wqkv, const float* __restrict__ wproj,
                       _Float16* __restrict__ wq16, _Float16* __restrict__ wp16,
                       float* __restrict__ nq, float* __restrict__ nk) {
  int i = blockIdx.x * 256 + threadIdx.x;
  if (i < O3 * DIM)  wq16[i] = (_Float16)wqkv[i];
  if (i < DIM * DIM) wp16[i] = (_Float16)wproj[i];
  if (i < B * HEADS * H) { nq[i] = 0.f; nk[i] = 0.f; }
}

// ---------------------------------------------------------------------------
// K1: x [B][C][P] f32 -> XT [B][P][C] f16 via 32x32 LDS tile transpose
// ---------------------------------------------------------------------------
__global__ void k_xt(const float* __restrict__ x, _Float16* __restrict__ xt) {
  __shared__ float t[32][33];
  int blk = blockIdx.x;
  int pT = blk % (P / 32); blk /= (P / 32);
  int cT = blk % (DIM / 32); blk /= (DIM / 32);
  int b  = blk;
  int c0 = cT * 32, p0 = pT * 32;
  const float* xb = x + (size_t)b * DIM * P;
  int col = threadIdx.x & 31, r0 = threadIdx.x >> 5;
#pragma unroll
  for (int rr = r0; rr < 32; rr += 8)
    t[rr][col] = xb[(size_t)(c0 + rr) * P + p0 + col];
  __syncthreads();
  _Float16* xtb = xt + ((size_t)b * P + p0) * DIM + c0;
#pragma unroll
  for (int rr = r0; rr < 32; rr += 8)
    xtb[(size_t)rr * DIM + col] = (_Float16)t[col][rr];
}

// ---------------------------------------------------------------------------
// K2: qkv 1x1 conv as GEMM  Wq(576x192) @ XT[b]^T -> qkv0[b][o][p] f16
//     A tile (32x192 = 12KB, contiguous) async-staged into LDS per block.
// ---------------------------------------------------------------------------
__global__ void k_gemm_qkv(const _Float16* __restrict__ wq, const _Float16* __restrict__ xt,
                           _Float16* __restrict__ qkv0) {
  __shared__ _Float16 As[32 * DIM];
  unsigned tid = threadIdx.x, lane = tid & 31, wave = tid >> 5;
  int nb = blockIdx.x % ((P / 32) / 8);              // 64
  int m2 = (blockIdx.x / 64) % (O3 / 32);            // 18
  int b  = blockIdx.x / (64 * (O3 / 32));
  int n2 = nb * 8 + wave;
  stage_A_async(wq + (size_t)m2 * 32 * DIM, As, 32 * DIM * 2, tid);
  const _Float16* BT = xt + (size_t)b * P * DIM;
  v8f acc[2][2];
  gemm_block_2x2_sA(As, DIM, BT, DIM, n2 * 32, DIM, lane, acc);
  _Float16* out = qkv0 + (size_t)b * O3 * P;
  int rowoff = (lane >> 4) ? 8 : 0, l15 = lane & 15;
#pragma unroll
  for (int ti = 0; ti < 2; ++ti)
#pragma unroll
    for (int tj = 0; tj < 2; ++tj) {
      int mr = m2 * 32 + ti * 16 + rowoff;
      int p  = n2 * 32 + tj * 16 + l15;
#pragma unroll
      for (int r = 0; r < 8; ++r) out[(size_t)(mr + r) * P + p] = (_Float16)acc[ti][tj][r];
    }
}

// ---------------------------------------------------------------------------
// K3: depthwise 3x3 + head-permute scatter + row sum-of-squares (q,k)
//   q -> qh[bh][h][f], k -> kh[bh][h][f], v -> vT[bh][f][h]; f = ch*W + w
// ---------------------------------------------------------------------------
__global__ void k_dw(const _Float16* __restrict__ qkv0, const float* __restrict__ wdw,
                     _Float16* __restrict__ qh, _Float16* __restrict__ kh,
                     _Float16* __restrict__ vT,
                     float* __restrict__ nq, float* __restrict__ nk) {
  __shared__ float ssq[2];
  int tid = threadIdx.x;
  int blk = blockIdx.x;
  int pb  = (blk & 63) << 8; blk >>= 6;     // 64 blocks of 256 pixels per channel
  int o   = blk % O3;
  int b   = blk / O3;
  int p = pb + tid, h = p >> 7, w = p & 127;
  const _Float16* in = qkv0 + ((size_t)b * O3 + o) * P;
  const float* wt = wdw + o * 9;
  float acc = 0.f;
#pragma unroll
  for (int dy = -1; dy <= 1; ++dy) {
    int hh = h + dy;
    if ((unsigned)hh < (unsigned)H) {
#pragma unroll
      for (int dx = -1; dx <= 1; ++dx) {
        int ww = w + dx;
        if ((unsigned)ww < (unsigned)W)
          acc += wt[(dy + 1) * 3 + (dx + 1)] * (float)in[hh * W + ww];
      }
    }
  }
  int kind = o / DIM;           // 0=q 1=k 2=v  (uniform per block)
  int oc = o % DIM, head = oc / CH, ch = oc % CH;
  size_t row = ((size_t)b * HEADS + head) * H + h;
  size_t f   = (size_t)ch * W + w;
  _Float16 hv = (_Float16)acc;
  if (kind == 2) {
    vT[(((size_t)b * HEADS + head) * F + f) * H + h] = hv;
  } else {
    ((kind == 0) ? qh : kh)[row * F + f] = hv;
    if (tid < 2) ssq[tid] = 0.f;
    __syncthreads();
    atomicAdd(&ssq[tid >> 7], acc * acc);   // ds_add_f32
    __syncthreads();
    if ((tid & 127) == 0)
      atomicAdd(&((kind == 0) ? nq : nk)[row], ssq[tid >> 7]);
  }
}

// ---------------------------------------------------------------------------
// K4: raw attention scores  Q[bh](128xF) @ K[bh]^T -> sraw f32 [bh][128][128]
//     (K=6144: pure streaming GEMM with global_prefetch_b8)
// ---------------------------------------------------------------------------
__global__ void k_scores(const _Float16* __restrict__ qh, const _Float16* __restrict__ kh,
                         float* __restrict__ sraw) {
  unsigned lane = threadIdx.x & 31, wave = threadIdx.x >> 5;
  size_t tile = (size_t)blockIdx.x * 8 + wave;
  int tn2 = (int)(tile & 3); tile >>= 2;
  int tm2 = (int)(tile & 3); tile >>= 2;
  int bh = (int)tile;
  const _Float16* A  = qh + (size_t)bh * H * F;
  const _Float16* BT = kh + (size_t)bh * H * F;
  v8f acc[2][2];
  gemm_block_2x2(A, F, BT, F, tm2 * 32, tn2 * 32, F, lane, acc);
  float* out = sraw + (size_t)bh * H * H;
  int rowoff = (lane >> 4) ? 8 : 0, l15 = lane & 15;
#pragma unroll
  for (int ti = 0; ti < 2; ++ti)
#pragma unroll
    for (int tj = 0; tj < 2; ++tj) {
      int i = tm2 * 32 + ti * 16 + rowoff;
      int j = tn2 * 32 + tj * 16 + l15;
#pragma unroll
      for (int r = 0; r < 8; ++r) out[(size_t)(i + r) * H + j] = acc[ti][tj][r];
    }
}

// ---------------------------------------------------------------------------
// K5: fused l2-norm scaling + temperature + softmax -> P f16 [bh][i][j]
// ---------------------------------------------------------------------------
__global__ void k_softmax(const float* __restrict__ sraw, const float* __restrict__ nq,
                          const float* __restrict__ nk, const float* __restrict__ temp,
                          _Float16* __restrict__ Pm) {
  __shared__ float red[4];
  int row  = blockIdx.x;                 // (b*HEADS+head)*H + i
  int bh   = row >> 7;
  int head = bh & (HEADS - 1);
  int tid  = threadIdx.x;                // j in [0,128)
  float t  = temp[head];
  float qn = fmaxf(sqrtf(nq[row]), 1e-12f);
  float kn = fmaxf(sqrtf(nk[(size_t)bh * H + tid]), 1e-12f);
  float s  = sraw[(size_t)row * H + tid] * t / (qn * kn);
  float m = s;
#pragma unroll
  for (int off = 16; off; off >>= 1) m = fmaxf(m, __shfl_xor(m, off, 32));
  if ((tid & 31) == 0) red[tid >> 5] = m;
  __syncthreads();
  m = fmaxf(fmaxf(red[0], red[1]), fmaxf(red[2], red[3]));
  __syncthreads();
  float e = __expf(s - m);
  float sum = e;
#pragma unroll
  for (int off = 16; off; off >>= 1) sum += __shfl_xor(sum, off, 32);
  if ((tid & 31) == 0) red[tid >> 5] = sum;
  __syncthreads();
  sum = red[0] + red[1] + red[2] + red[3];
  Pm[(size_t)row * H + tid] = (_Float16)(e / sum);
}

// ---------------------------------------------------------------------------
// K6: out = P(128x128) @ V ; A tile (32x128 = 8KB) async-staged into LDS.
//     Written directly permuted: po[b][p][cin] f16
// ---------------------------------------------------------------------------
__global__ void k_av(const _Float16* __restrict__ Pm, const _Float16* __restrict__ vT,
                     _Float16* __restrict__ po) {
  __shared__ _Float16 As[32 * H];
  unsigned tid = threadIdx.x, lane = tid & 31, wave = tid >> 5;
  int nb  = blockIdx.x % ((F / 32) / 8);             // 24
  int tm2 = (blockIdx.x / 24) & 3;
  int bh  = blockIdx.x / 96;
  int tn2 = nb * 8 + wave;
  int b = bh >> 2, head = bh & (HEADS - 1);
  stage_A_async(Pm + (size_t)bh * H * H + (size_t)tm2 * 32 * H, As, 32 * H * 2, tid);
  const _Float16* BT = vT + (size_t)bh * F * H;      // [f][j]
  v8f acc[2][2];
  gemm_block_2x2_sA(As, H, BT, H, tn2 * 32, H, lane, acc);
  int rowoff = (lane >> 4) ? 8 : 0, l15 = lane & 15;
#pragma unroll
  for (int ti = 0; ti < 2; ++ti)
#pragma unroll
    for (int tj = 0; tj < 2; ++tj) {
      int i = tm2 * 32 + ti * 16 + rowoff;
      int f = tn2 * 32 + tj * 16 + l15;
      int ch = f >> 7, w = f & 127;
      _Float16* out = po + (size_t)b * P * DIM + (size_t)w * DIM + head * CH + ch;
#pragma unroll
      for (int r = 0; r < 8; ++r) out[(size_t)(i + r) * W * DIM] = (_Float16)acc[ti][tj][r];  // p=(i+r)*W+w
    }
}

// ---------------------------------------------------------------------------
// K7: proj 1x1 conv  Wp(192x192) @ po[b]^T -> y f32 [b][cout][p]
//     A tile (32x192 = 12KB) async-staged into LDS per block.
// ---------------------------------------------------------------------------
__global__ void k_proj(const _Float16* __restrict__ wp, const _Float16* __restrict__ po,
                       float* __restrict__ y) {
  __shared__ _Float16 As[32 * DIM];
  unsigned tid = threadIdx.x, lane = tid & 31, wave = tid >> 5;
  int nb = blockIdx.x % ((P / 32) / 8);              // 64
  int m2 = (blockIdx.x / 64) % (DIM / 32);           // 6
  int b  = blockIdx.x / (64 * (DIM / 32));
  int n2 = nb * 8 + wave;
  stage_A_async(wp + (size_t)m2 * 32 * DIM, As, 32 * DIM * 2, tid);
  const _Float16* BT = po + (size_t)b * P * DIM;
  v8f acc[2][2];
  gemm_block_2x2_sA(As, DIM, BT, DIM, n2 * 32, DIM, lane, acc);
  float* out = y + (size_t)b * DIM * P;
  int rowoff = (lane >> 4) ? 8 : 0, l15 = lane & 15;
#pragma unroll
  for (int ti = 0; ti < 2; ++ti)
#pragma unroll
    for (int tj = 0; tj < 2; ++tj) {
      int co = m2 * 32 + ti * 16 + rowoff;
      int p  = n2 * 32 + tj * 16 + l15;
#pragma unroll
      for (int r = 0; r < 8; ++r) out[(size_t)(co + r) * P + p] = acc[ti][tj][r];
    }
}

// ---------------------------------------------------------------------------
// Compile-probe kernel (NOT launched): TDM tensor load + cluster builtins,
// kept so the assembler continues to validate these paths for future use.
// ---------------------------------------------------------------------------
__global__ void k_cdna5_probe(unsigned long long* __restrict__ sink,
                              const unsigned long long* __restrict__ src) {
  __shared__ long long lds[512];
  unsigned ldsAddr = (unsigned)(unsigned long long)(&lds[0]) + threadIdx.x * 16u;
  unsigned long long ga = (unsigned long long)src + threadIdx.x * 16ull;
  asm volatile("global_load_async_to_lds_b128 %0, %1, off" :: "v"(ldsAddr), "v"(ga) : "memory");
  asm volatile("s_wait_asynccnt 0x0" ::: "memory");
  v4u g0 = {0u, 0u, 0u, (2u << 30)};          // null TDM descriptor, type=2
  v8u g1 = {0u, 0u, 0u, 0u, 0u, 0u, 0u, 0u};
  asm volatile("tensor_load_to_lds %0, %1" :: "s"(g0), "s"(g1) : "memory");
  asm volatile("s_wait_tensorcnt 0x0" ::: "memory");
  __builtin_amdgcn_s_cluster_barrier();
  sink[threadIdx.x] = lds[threadIdx.x] + (unsigned long long)__builtin_amdgcn_cluster_id_x();
}

// ---------------------------------------------------------------------------
extern "C" void kernel_launch(void* const* d_in, const int* in_sizes, int n_in,
                              void* d_out, int out_size, void* d_ws, size_t ws_size,
                              hipStream_t stream) {
  (void)in_sizes; (void)n_in; (void)out_size; (void)ws_size;
  const float* x     = (const float*)d_in[0];
  const float* wqkv  = (const float*)d_in[1];
  const float* wdw   = (const float*)d_in[2];
  const float* wproj = (const float*)d_in[3];
  const float* temp  = (const float*)d_in[4];
  float* y = (float*)d_out;
  char* ws = (char*)d_ws;

  _Float16* XT   = (_Float16*)(ws + OFF_XT);
  _Float16* WQ   = (_Float16*)(ws + OFF_WQ);
  _Float16* WP   = (_Float16*)(ws + OFF_WP);
  float*    NQ   = (float*)   (ws + OFF_NQ);
  float*    NK   = (float*)   (ws + OFF_NK);
  _Float16* QKV0 = (_Float16*)(ws + OFF_QKV0);
  _Float16* QH   = (_Float16*)(ws + OFF_QH);
  _Float16* KH   = (_Float16*)(ws + OFF_KH);
  _Float16* VT   = (_Float16*)(ws + OFF_VT);
  float*    SRAW = (float*)   (ws + OFF_SRAW);
  _Float16* PA   = (_Float16*)(ws + OFF_PA);
  _Float16* PO   = (_Float16*)(ws + OFF_PO);

  k_prep<<<(O3 * DIM + 255) / 256, 256, 0, stream>>>(wqkv, wproj, WQ, WP, NQ, NK);
  k_xt<<<B * (DIM / 32) * (P / 32), 256, 0, stream>>>(x, XT);
  k_gemm_qkv<<<B * (O3 / 32) * (P / 32) / 8, 256, 0, stream>>>(WQ, XT, QKV0);
  k_dw<<<B * O3 * (P / 256), 256, 0, stream>>>(QKV0, wdw, QH, KH, VT, NQ, NK);
  k_scores<<<B * HEADS * (H / 32) * (H / 32) / 8, 256, 0, stream>>>(QH, KH, SRAW);
  k_softmax<<<B * HEADS * H, 128, 0, stream>>>(SRAW, NQ, NK, temp, PA);
  k_av<<<B * HEADS * (H / 32) * (F / 32) / 8, 256, 0, stream>>>(PA, VT, PO);
  k_proj<<<B * (DIM / 32) * (P / 32) / 8, 256, 0, stream>>>(WP, PO, y);
}
